// memory_Loss_36593121362040
// MI455X (gfx1250) — matrix operand order repaired and checked by reading, not compiled, not run on hardware
//
#include <hip/hip_runtime.h>

#define FEATURE_DIM 1024
#define CLASS_NUM   8192
#define NTOT        16384          // 2 * CLASS_NUM
#define INV_TEMP    2.0f           // 1 / 0.5
#define LDS_ROW     1032           // 1024 bf16 + 8 bf16 pad (16B) -> 2064B row stride

typedef __attribute__((ext_vector_type(16))) __bf16        v16bf;
typedef __attribute__((ext_vector_type(8)))  float         v8f;
typedef __attribute__((ext_vector_type(4)))  unsigned int  u32x4;

union Frag32 {                      // 32 bytes per lane = 16 bf16 = A or B fragment
    u32x4 q[2];
    v16bf v;
};

__device__ __forceinline__ unsigned short f2bf(float f) {
    unsigned int u = __float_as_uint(f);
    u += 0x7fffu + ((u >> 16) & 1u);          // round-to-nearest-even
    return (unsigned short)(u >> 16);
}

// ---------------------------------------------------------------------------
// Kernel 1: V[r, d] = concat(h_i, h_j)[d, r] as bf16   (V is [NTOT, FEATURE_DIM])
// ---------------------------------------------------------------------------
__global__ __launch_bounds__(256) void transpose_cast_kernel(
    const float* __restrict__ h_i, const float* __restrict__ h_j,
    unsigned short* __restrict__ V)
{
    __shared__ float tile[32][33];
    const int cb = blockIdx.x * 32;           // class (N) base; 8192 % 32 == 0
    const int db = blockIdx.y * 32;           // feature (D) base
    const int tx = threadIdx.x;               // 0..31
    const int ty = threadIdx.y;               // 0..7
    const float* src = (cb < CLASS_NUM) ? h_i : h_j;
    const int csrc   = (cb < CLASS_NUM) ? cb : (cb - CLASS_NUM);
#pragma unroll
    for (int yy = 0; yy < 4; ++yy) {
        const int d = db + ty + yy * 8;
        tile[ty + yy * 8][tx] = src[(size_t)d * CLASS_NUM + csrc + tx];
    }
    __syncthreads();
#pragma unroll
    for (int yy = 0; yy < 4; ++yy) {
        const int r = cb + ty + yy * 8;
        const int d = db + tx;
        V[(size_t)r * FEATURE_DIM + d] = f2bf(tile[tx][ty + yy * 8]);
    }
}

// ---------------------------------------------------------------------------
// Kernel 2: fused Gram-matrix + streaming logsumexp NT-Xent.
// One wave owns 16 rows of sim; sweeps all 16384 columns in 128-wide blocks.
// A fragments come from LDS (this WG's 128 rows of V), B straight from L2.
// ---------------------------------------------------------------------------
__global__ __launch_bounds__(256) void ntxent_wmma_kernel(
    const unsigned short* __restrict__ V, float* __restrict__ partials)
{
    extern __shared__ unsigned short sA[];    // [128][LDS_ROW] bf16 (padded rows)
    const int tid  = threadIdx.x;
    const int lane = tid & 31;
    const int wave = tid >> 5;
    const int cl   = lane & 15;               // column-within-tile / A-row select
    const int hi   = lane >> 4;               // half-wave select
    const int rowBlock = blockIdx.x * 128;
    const int i0   = rowBlock + wave * 16;

    // Stage 128 rows x 1024 bf16 into LDS with per-row padding (bank rotation).
    {
        const u32x4* src = (const u32x4*)(V + (size_t)rowBlock * FEATURE_DIM);
        u32x4* dst = (u32x4*)sA;
        for (int i = tid; i < 128 * FEATURE_DIM / 8; i += 256) {
            const int row = i >> 7;           // 128 u32x4 per source row
            const int off = i & 127;
            dst[row * (LDS_ROW / 8) + off] = src[i];   // LDS_ROW/8 == 129
        }
    }
    __syncthreads();

    float m[8], s[8], pos[8];
#pragma unroll
    for (int v = 0; v < 8; ++v) {
        m[v] = -__builtin_inff(); s[v] = 0.0f; pos[v] = -__builtin_inff();
    }

    const int rbase = i0 + 8 * hi;            // C/D layout: row = rbase + vgpr index
    const u32x4* aRow = (const u32x4*)(sA + (size_t)(wave * 16 + cl) * LDS_ROW);

#pragma unroll 1
    for (int jb = 0; jb < NTOT; jb += 128) {
        v8f acc[8];
#pragma unroll
        for (int t = 0; t < 8; ++t)
#pragma unroll
            for (int v = 0; v < 8; ++v) acc[t][v] = 0.0f;

#pragma unroll 2
        for (int k = 0; k < FEATURE_DIM; k += 32) {
            // A 16x32 bf16 fragment: lane<16 K={k..k+15},
            // lane>=16 K={k+16..k+31} (contiguous, matches B).
            Frag32 a;
            a.q[0] = aRow[(k + hi * 16) >> 3];
            a.q[1] = aRow[(k + hi * 16 + 8) >> 3];
#pragma unroll
            for (int t = 0; t < 8; ++t) {
                const int col = jb + t * 16 + cl;
                const u32x4* bp =
                    (const u32x4*)(V + (size_t)col * FEATURE_DIM + k + hi * 16);
                Frag32 b;
                b.q[0] = bp[0];               // col-major B: 16 contiguous K per lane
                b.q[1] = bp[1];
                acc[t] = __builtin_amdgcn_wmma_f32_16x16x32_bf16(
                    false, a.v, false, b.v, (short)0, acc[t], false, false);
            }
        }

        // Streaming logsumexp update; exclude diagonal, capture positive pair.
#pragma unroll
        for (int t = 0; t < 8; ++t) {
            const int c = jb + t * 16 + cl;
#pragma unroll
            for (int v = 0; v < 8; ++v) {
                const int r = rbase + v;
                if (c == r) continue;                      // mask diagonal only
                const float x = acc[t][v] * INV_TEMP;
                if (c == ((r + CLASS_NUM) & (NTOT - 1))) pos[v] = x;
                const float nm = fmaxf(m[v], x);
                s[v] = s[v] * __expf(m[v] - nm) + __expf(x - nm);
                m[v] = nm;
            }
        }
    }

    // Butterfly lse-merge across the 16-lane column group (wave32 shuffles).
#pragma unroll
    for (int b = 0; b < 4; ++b) {
        const int mask = 1 << b;
#pragma unroll
        for (int v = 0; v < 8; ++v) {
            const float om = __shfl_xor(m[v], mask, 32);
            const float os = __shfl_xor(s[v], mask, 32);
            const float op = __shfl_xor(pos[v], mask, 32);
            const float nm = fmaxf(m[v], om);
            s[v] = s[v] * __expf(m[v] - nm) + os * __expf(om - nm);
            m[v] = nm;
            pos[v] = fmaxf(pos[v], op);
        }
    }

    float lsum = 0.0f;
#pragma unroll
    for (int v = 0; v < 8; ++v) lsum += m[v] + __logf(s[v]) - pos[v];
    lsum += __shfl_xor(lsum, 16, 32);          // rows 0-7 half + rows 8-15 half
    if (lane == 0) partials[blockIdx.x * 8 + wave] = lsum;
}

// ---------------------------------------------------------------------------
// Kernel 3: deterministic tree reduction of 1024 per-wave partials.
// ---------------------------------------------------------------------------
__global__ __launch_bounds__(256) void reduce_kernel(
    const float* __restrict__ partials, float* __restrict__ out)
{
    __shared__ float sh[256];
    float a = 0.0f;
    for (int i = threadIdx.x; i < 1024; i += 256) a += partials[i];
    sh[threadIdx.x] = a;
    __syncthreads();
    for (int st = 128; st > 0; st >>= 1) {
        if ((int)threadIdx.x < st) sh[threadIdx.x] += sh[threadIdx.x + st];
        __syncthreads();
    }
    if (threadIdx.x == 0) out[0] = sh[0] * (1.0f / (float)NTOT);
}

extern "C" void kernel_launch(void* const* d_in, const int* in_sizes, int n_in,
                              void* d_out, int out_size, void* d_ws, size_t ws_size,
                              hipStream_t stream) {
    const float* h_i = (const float*)d_in[0];
    const float* h_j = (const float*)d_in[1];

    unsigned short* V = (unsigned short*)d_ws;                       // 32 MB bf16
    float* partials = (float*)((char*)d_ws +
                               (size_t)NTOT * FEATURE_DIM * sizeof(unsigned short));

    dim3 gT(NTOT / 32, FEATURE_DIM / 32);
    dim3 bT(32, 8);
    transpose_cast_kernel<<<gT, bT, 0, stream>>>(h_i, h_j, V);

    const size_t ldsBytes = (size_t)128 * LDS_ROW * sizeof(unsigned short); // 258 KB
    ntxent_wmma_kernel<<<NTOT / 128, 256, ldsBytes, stream>>>(V, partials);

    reduce_kernel<<<1, 256, 0, stream>>>(partials, (float*)d_out);
}
